// SuperPointMatching_80169859547182
// MI455X (gfx1250) — compile-verified, matrix operand order checked by reading, not run on hardware
//
#include <hip/hip_runtime.h>

typedef __attribute__((ext_vector_type(16))) _Float16 v16h;
typedef __attribute__((ext_vector_type(8)))  float    v8f;
typedef __attribute__((ext_vector_type(2)))  _Float16 v2h;

#define NPTS 8192
#define DDIM 512
#define CDIM 128
#define LDSS 40                 // LDS row stride in halves: 80B, 16B-aligned, conflict-free b128
#define TILEH (128 * LDSS)      // halves per tile buffer (10240 B)
#define CAND_MAX 4096

// ---- workspace layout (bytes) ----
#define WS_ROWSUM  0u                      // f32[8192]
#define WS_COLSUM  32768u                  // f32[8192]
#define WS_TMAX    65536u                  // f32[4096] per-tile maxima
#define WS_COUNTER 81920u                  // u32
#define WS_THR     81924u                  // u32 (float bits of threshold)
#define WS_CAND    98304u                  // u64[4096]
#define WS_REFH    131072ull               // f16[8192*512] = 8MB
#define WS_SRCH    (WS_REFH + 8ull*1024*1024)
#define WS_CAS0T   (WS_SRCH + 8ull*1024*1024)   // f16[8192*128] transposed [n][c]
#define WS_CAS1T   (WS_CAS0T + 2ull*1024*1024)

// ------------------------------------------------------------------
__global__ void zero_u32(unsigned* p, int n) {
    int i = blockIdx.x * 256 + threadIdx.x;
    if (i < n) p[i] = 0u;
}

__global__ void convert_feats(const float* __restrict__ a, const float* __restrict__ b,
                              _Float16* __restrict__ ah, _Float16* __restrict__ bh) {
    int i = blockIdx.x * 256 + threadIdx.x;   // over (8192*512)/2
    float2 fa = reinterpret_cast<const float2*>(a)[i];
    float2 fb = reinterpret_cast<const float2*>(b)[i];
    v2h ha; ha[0] = (_Float16)fa.x; ha[1] = (_Float16)fa.y;
    v2h hb; hb[0] = (_Float16)fb.x; hb[1] = (_Float16)fb.y;
    reinterpret_cast<v2h*>(ah)[i] = ha;
    reinterpret_cast<v2h*>(bh)[i] = hb;
}

__global__ void convert_cas(const float* __restrict__ c0, const float* __restrict__ c1,
                            _Float16* __restrict__ t0, _Float16* __restrict__ t1) {
    int idx = blockIdx.x * 256 + threadIdx.x; // over 8192*128
    int n = idx >> 7, c = idx & 127;
    t0[idx] = (_Float16)c0[c * NPTS + n];
    t1[idx] = (_Float16)c1[c * NPTS + n];
}

// ------------------------------------------------------------------
// async 32B (one lane-chunk) global -> LDS, GVS mode; offset advances BOTH addresses
__device__ __forceinline__ void fill_async(const _Float16* __restrict__ g, unsigned gByteOff,
                                           unsigned ldsByteOff) {
    asm volatile(
        "global_load_async_to_lds_b128 %0, %1, %2\n\t"
        "global_load_async_to_lds_b128 %0, %1, %2 offset:16"
        :
        : "v"(ldsByteOff), "v"(gByteOff), "s"((unsigned long long)(size_t)g)
        : "memory");
}

template <int N>
__device__ __forceinline__ void wait_async() {
    asm volatile("s_wait_asynccnt %0" :: "i"(N) : "memory");
}

// per-lane A/B fragment: row/col = (lane&15), K-halves {8b..8b+7} U {16+8b..16+8b+7}, b=lane>>4
__device__ __forceinline__ v16h frag_load(const _Float16* lds, int row, int lane) {
    const _Float16* p = lds + row * LDSS + ((lane >> 4) << 3);
    v16h f;
#pragma unroll
    for (int j = 0; j < 8; ++j) f[j] = p[j];
#pragma unroll
    for (int j = 0; j < 8; ++j) f[j + 8] = p[16 + j];
    return f;
}

// double-buffered, async-fed 128x128 tile GEMM; ldsT = 4 buffers [A0|A1|B0|B1]
template <int NK>
__device__ __forceinline__ void gemm_tile(const _Float16* __restrict__ gA,
                                          const _Float16* __restrict__ gB, int ldg,
                                          int rowA0, int rowB0,
                                          _Float16* ldsT, v8f acc[2][4], int tid) {
    int lane = tid & 31, wave = tid >> 5;
    int wRow = (wave >> 1) * 32, wCol = (wave & 1) * 64;
    int r = tid >> 1;                 // 0..127 rows
    int h = (tid & 1) << 4;           // 0 / 16 halves
    unsigned offA = (unsigned)(((rowA0 + r) * ldg + h) * 2);   // byte offsets
    unsigned offB = (unsigned)(((rowB0 + r) * ldg + h) * 2);
    unsigned ldsRow = (unsigned)((r * LDSS + h) * 2);
    unsigned ldsBase = (unsigned)(size_t)ldsT;                 // low 32 bits = LDS addr
    unsigned la0 = ldsBase + ldsRow;
    unsigned la1 = la0 + 1u * TILEH * 2u;
    unsigned lb0 = la0 + 2u * TILEH * 2u;
    unsigned lb1 = la0 + 3u * TILEH * 2u;

    // prologue: step 0 -> buffer 0
    fill_async(gA, offA, la0);
    fill_async(gB, offB, lb0);

    for (int kt = 0; kt < NK; ++kt) {
        if (kt + 1 < NK) {              // prefetch next step into other buffer
            unsigned kb = (unsigned)((kt + 1) * 64);  // 32 halves = 64B per K-step
            if ((kt + 1) & 1) { fill_async(gA, offA + kb, la1); fill_async(gB, offB + kb, lb1); }
            else              { fill_async(gA, offA + kb, la0); fill_async(gB, offB + kb, lb0); }
            wait_async<4>();            // oldest 4 (current step) complete, in-order
        } else {
            wait_async<0>();
        }
        __syncthreads();                // all waves' chunks of this step visible
        const _Float16* bufA = ldsT + (kt & 1) * TILEH;
        const _Float16* bufB = ldsT + (2 + (kt & 1)) * TILEH;
        v16h a[2], b[4];
#pragma unroll
        for (int wr = 0; wr < 2; ++wr) a[wr] = frag_load(bufA, wRow + wr * 16 + (lane & 15), lane);
#pragma unroll
        for (int wc = 0; wc < 4; ++wc) b[wc] = frag_load(bufB, wCol + wc * 16 + (lane & 15), lane);
#pragma unroll
        for (int wr = 0; wr < 2; ++wr)
#pragma unroll
            for (int wc = 0; wc < 4; ++wc)
                acc[wr][wc] = __builtin_amdgcn_wmma_f32_16x16x32_f16(
                    false, a[wr], false, b[wc], (short)0, acc[wr][wc], false, false);
        __syncthreads();                // reads done before buffer reused
    }
}

// ------------------------------------------------------------------
// MODE 1: row/col sums of exp(2*inner-2)
// MODE 2: per-tile max of final score
// MODE 3: collect candidates with score bits >= threshold
template <int MODE>
__global__ __launch_bounds__(256) void tile_kernel(
    const _Float16* __restrict__ refH, const _Float16* __restrict__ srcH,
    const _Float16* __restrict__ cas0T, const _Float16* __restrict__ cas1T,
    float* __restrict__ rowsum, float* __restrict__ colsum,
    float* __restrict__ tmax, unsigned* __restrict__ counter,
    const unsigned* __restrict__ thrBits, unsigned long long* __restrict__ cand) {
    __shared__ __align__(16) _Float16 ldsT[4 * TILEH];   // A0|A1|B0|B1, 40KB
    __shared__ float rsum_l[128], csum_l[128];
    __shared__ float wredu[8];

    int tid = threadIdx.x;
    int lane = tid & 31, wave = tid >> 5;
    int wRow = (wave >> 1) * 32, wCol = (wave & 1) * 64;
    int tileN = blockIdx.y * 128, tileM = blockIdx.x * 128;

    v8f acc[2][4], ov[2][4];
#pragma unroll
    for (int wr = 0; wr < 2; ++wr)
#pragma unroll
        for (int wc = 0; wc < 4; ++wc)
#pragma unroll
            for (int e = 0; e < 8; ++e) { acc[wr][wc][e] = 0.0f; ov[wr][wc][e] = 0.0f; }

    if (MODE == 1 && tid < 128) { rsum_l[tid] = 0.0f; csum_l[tid] = 0.0f; }

    // features GEMM: D=512, 16 K-steps
    gemm_tile<16>(refH, srcH, DDIM, tileN, tileM, ldsT, acc, tid);

    if (MODE != 1) {
        // overlap GEMM: C=128, 4 K-steps, reuses LDS buffers
        gemm_tile<4>(cas0T, cas1T, CDIM, tileN, tileM, ldsT, ov, tid);
        if (tid < 128) rsum_l[tid] = __builtin_amdgcn_rcpf(rowsum[tileN + tid]);
        else           csum_l[tid - 128] = __builtin_amdgcn_rcpf(colsum[tileM + tid - 128]);
        __syncthreads();
    }

    if (MODE == 1) {
        float rowAcc[16];
#pragma unroll
        for (int i = 0; i < 16; ++i) rowAcc[i] = 0.0f;
        float colAcc[4] = {0.f, 0.f, 0.f, 0.f};
#pragma unroll
        for (int wr = 0; wr < 2; ++wr)
#pragma unroll
            for (int wc = 0; wc < 4; ++wc)
#pragma unroll
                for (int g = 0; g < 8; ++g) {
                    float e = __expf(2.0f * acc[wr][wc][g] - 2.0f);
                    rowAcc[wr * 8 + g] += e;
                    colAcc[wc] += e;
                }
#pragma unroll
        for (int i = 0; i < 16; ++i) {  // 16-lane horizontal sums
            float v = rowAcc[i];
            v += __shfl_xor(v, 1); v += __shfl_xor(v, 2);
            v += __shfl_xor(v, 4); v += __shfl_xor(v, 8);
            if ((lane & 15) == 0)
                atomicAdd(&rsum_l[wRow + (i >> 3) * 16 + (i & 7) + 8 * (lane >> 4)], v);
        }
#pragma unroll
        for (int wc = 0; wc < 4; ++wc)
            atomicAdd(&csum_l[wCol + wc * 16 + (lane & 15)], colAcc[wc]);
        __syncthreads();
        if (tid < 128) atomicAdd(&rowsum[tileN + tid], rsum_l[tid]);
        else           atomicAdd(&colsum[tileM + tid - 128], csum_l[tid - 128]);
    } else if (MODE == 2) {
        float m = 0.0f;
#pragma unroll
        for (int wr = 0; wr < 2; ++wr)
#pragma unroll
            for (int wc = 0; wc < 4; ++wc)
#pragma unroll
                for (int g = 0; g < 8; ++g) {
                    int nl = wRow + wr * 16 + g + 8 * (lane >> 4);
                    int ml = wCol + wc * 16 + (lane & 15);
                    float e = __expf(2.0f * acc[wr][wc][g] - 2.0f);
                    float v = (e * rsum_l[nl]) * (e * csum_l[ml]) * ov[wr][wc][g];
                    m = fmaxf(m, v);
                }
        m = fmaxf(m, __shfl_xor(m, 1));  m = fmaxf(m, __shfl_xor(m, 2));
        m = fmaxf(m, __shfl_xor(m, 4));  m = fmaxf(m, __shfl_xor(m, 8));
        m = fmaxf(m, __shfl_xor(m, 16));
        if (lane == 0) wredu[wave] = m;
        __syncthreads();
        if (tid == 0) {
            float t = wredu[0];
#pragma unroll
            for (int w = 1; w < 8; ++w) t = fmaxf(t, wredu[w]);
            tmax[blockIdx.y * (NPTS / 128) + blockIdx.x] = t;
        }
    } else {
        unsigned thr = *thrBits;
#pragma unroll
        for (int wr = 0; wr < 2; ++wr)
#pragma unroll
            for (int wc = 0; wc < 4; ++wc)
#pragma unroll
                for (int g = 0; g < 8; ++g) {
                    int nl = wRow + wr * 16 + g + 8 * (lane >> 4);
                    int ml = wCol + wc * 16 + (lane & 15);
                    float e = __expf(2.0f * acc[wr][wc][g] - 2.0f);
                    float v = (e * rsum_l[nl]) * (e * csum_l[ml]) * ov[wr][wc][g];
                    unsigned bits = __float_as_uint(v);
                    if (bits >= thr) {
                        unsigned pos = atomicAdd(counter, 1u);
                        if (pos < CAND_MAX)
                            cand[pos] = ((unsigned long long)bits << 32) |
                                        (unsigned)((unsigned)(tileN + nl) * (unsigned)NPTS +
                                                   (unsigned)(tileM + ml));
                    }
                }
    }
}

// ------------------------------------------------------------------
// threshold = 256th-largest tile max (exact, via binary search on float bits)
__global__ __launch_bounds__(256) void select_threshold(const float* __restrict__ tmax,
                                                        unsigned* __restrict__ counter,
                                                        unsigned* __restrict__ thr) {
    __shared__ unsigned tm[4096];
    __shared__ unsigned red[256];
    __shared__ unsigned curS;
    int tid = threadIdx.x;
    for (int i = tid; i < 4096; i += 256) tm[i] = __float_as_uint(tmax[i]); // all >= 0
    if (tid == 0) curS = 0u;
    __syncthreads();
    for (int bit = 30; bit >= 0; --bit) {
        unsigned trial = curS | (1u << bit);
        unsigned c = 0;
        for (int i = tid; i < 4096; i += 256) c += (tm[i] >= trial) ? 1u : 0u;
        red[tid] = c;
        __syncthreads();
        for (int s = 128; s > 0; s >>= 1) {
            if (tid < s) red[tid] += red[tid + s];
            __syncthreads();
        }
        if (tid == 0 && red[0] >= 256u) curS = trial;
        __syncthreads();
    }
    if (tid == 0) { *thr = curS; *counter = 0u; }
}

__device__ __forceinline__ unsigned long long umax64(unsigned long long a, unsigned long long b) {
    return a > b ? a : b;
}

// exact top-256 extraction from <= CAND_MAX candidates (keys = score_bits<<32 | flat_idx)
__global__ __launch_bounds__(256) void final_topk(const unsigned long long* __restrict__ cand,
                                                  const unsigned* __restrict__ counter,
                                                  float* __restrict__ out) {
    __shared__ unsigned long long cs[CAND_MAX];
    __shared__ unsigned long long red[256];
    int tid = threadIdx.x;
    unsigned n = *counter; if (n > CAND_MAX) n = CAND_MAX;
    for (int i = tid; i < CAND_MAX; i += 256) cs[i] = (i < (int)n) ? cand[i] : 0ull;
    __syncthreads();
    for (int it = 0; it < 256; ++it) {
        unsigned long long best = 0ull;
        for (int i = tid; i < CAND_MAX; i += 256) best = umax64(best, cs[i]);
        red[tid] = best;
        __syncthreads();
        for (int s = 128; s > 0; s >>= 1) {
            if (tid < s) red[tid] = umax64(red[tid], red[tid + s]);
            __syncthreads();
        }
        unsigned long long w = red[0];
        for (int i = tid; i < CAND_MAX; i += 256)
            if (cs[i] == w) cs[i] = 0ull;          // keys unique (idx unique)
        if (tid == 0) {
            unsigned idx = (unsigned)(w & 0xffffffffu);
            out[it]       = (float)(idx / (unsigned)NPTS);        // ref index
            out[256 + it] = (float)(idx % (unsigned)NPTS);        // src index
            out[512 + it] = __uint_as_float((unsigned)(w >> 32)); // score
        }
        __syncthreads();
    }
}

// ------------------------------------------------------------------
extern "C" void kernel_launch(void* const* d_in, const int* in_sizes, int n_in,
                              void* d_out, int out_size, void* d_ws, size_t ws_size,
                              hipStream_t stream) {
    const float* ref  = (const float*)d_in[0];
    const float* src  = (const float*)d_in[1];
    const float* cas0 = (const float*)d_in[2];
    const float* cas1 = (const float*)d_in[3];
    (void)in_sizes; (void)n_in; (void)out_size; (void)ws_size;

    char* ws = (char*)d_ws;
    float*    rowsum  = (float*)(ws + WS_ROWSUM);
    float*    colsum  = (float*)(ws + WS_COLSUM);
    float*    tmax    = (float*)(ws + WS_TMAX);
    unsigned* counter = (unsigned*)(ws + WS_COUNTER);
    unsigned* thr     = (unsigned*)(ws + WS_THR);
    unsigned long long* cand = (unsigned long long*)(ws + WS_CAND);
    _Float16* refH  = (_Float16*)(ws + WS_REFH);
    _Float16* srcH  = (_Float16*)(ws + WS_SRCH);
    _Float16* cas0T = (_Float16*)(ws + WS_CAS0T);
    _Float16* cas1T = (_Float16*)(ws + WS_CAS1T);
    float* out = (float*)d_out;

    zero_u32<<<(20482 + 255) / 256, 256, 0, stream>>>((unsigned*)ws, 20482);
    convert_feats<<<(NPTS * DDIM / 2) / 256, 256, 0, stream>>>(ref, src, refH, srcH);
    convert_cas<<<(NPTS * CDIM) / 256, 256, 0, stream>>>(cas0, cas1, cas0T, cas1T);

    dim3 g(NPTS / 128, NPTS / 128);
    tile_kernel<1><<<g, 256, 0, stream>>>(refH, srcH, cas0T, cas1T, rowsum, colsum, tmax, counter, thr, cand);
    tile_kernel<2><<<g, 256, 0, stream>>>(refH, srcH, cas0T, cas1T, rowsum, colsum, tmax, counter, thr, cand);
    select_threshold<<<1, 256, 0, stream>>>(tmax, counter, thr);
    tile_kernel<3><<<g, 256, 0, stream>>>(refH, srcH, cas0T, cas1T, rowsum, colsum, tmax, counter, thr, cand);
    final_topk<<<1, 256, 0, stream>>>(cand, counter, out);
}
